// MultiHeadAttention_19112604467272
// MI455X (gfx1250) — matrix operand order repaired and checked by reading, not compiled, not run on hardware
//
#include <hip/hip_runtime.h>

#define B_   2
#define S_   2048
#define D_   1024
#define H_   16
#define DK_  64
#define NEGV -1.0e9f
#define SSCALE 0.18033688011112042f   // (1/sqrt(64)) * log2(e)

typedef __bf16 bf16;
typedef __attribute__((ext_vector_type(16))) __bf16 v16bf;
typedef __attribute__((ext_vector_type(8)))  __bf16 v8bf;
typedef __attribute__((ext_vector_type(8)))  float  v8f;

__device__ __forceinline__ v8f wmma_bf16(v16bf a, v16bf b, v8f c) {
  // v_wmma_f32_16x16x32_bf16 (neg_a, A, neg_b, B, c_mod, C, reuse_a, reuse_b)
  return __builtin_amdgcn_wmma_f32_16x16x32_bf16(false, a, false, b, (short)0, c, false, false);
}

// ---- fragment loaders (wave32 layouts per CDNA5 ISA 7.12.2) ----
// A 16x32 bf16: lane = M, element e -> K = 16*(e>>3) + 8*hl + (e&7)
// => two contiguous 8-elem (16B) chunks at kbase+8*hl and kbase+16+8*hl.
__device__ __forceinline__ v16bf load_a_bf16(const bf16* __restrict__ base, int ld, int kbase) {
  const int lane = threadIdx.x & 31;
  const int m = lane & 15, hl = lane >> 4;
  const bf16* p = base + (size_t)m * ld + kbase + 8 * hl;
  v8bf c0 = *(const v8bf*)(p);
  v8bf c1 = *(const v8bf*)(p + 16);
  v16bf a;
#pragma unroll
  for (int i = 0; i < 8; ++i) { a[i] = c0[i]; a[8 + i] = c1[i]; }
  return a;
}

__device__ __forceinline__ v16bf load_a_f32(const float* __restrict__ base, int ld, int kbase) {
  const int lane = threadIdx.x & 31;
  const int m = lane & 15, hl = lane >> 4;
  const float* p = base + (size_t)m * ld + kbase + 8 * hl;
  v8f c0 = *(const v8f*)(p);
  v8f c1 = *(const v8f*)(p + 16);
  v16bf a;
#pragma unroll
  for (int i = 0; i < 8; ++i) { a[i] = (bf16)c0[i]; a[8 + i] = (bf16)c1[i]; }
  return a;
}

// B 32x16: lane = N, element e -> K = 16*(e>>3) + 8*hl + (e&7) (same as A).
// Memory laid out mem[n][k] (B^T row-major): two 16B loads per lane.
__device__ __forceinline__ v16bf load_bt_bf16(const bf16* __restrict__ base, int ld, int n0, int kbase) {
  const int lane = threadIdx.x & 31;
  const int n = n0 + (lane & 15), hl = lane >> 4;
  const bf16* p = base + (size_t)n * ld + kbase + 8 * hl;
  v8bf c0 = *(const v8bf*)(p);
  v8bf c1 = *(const v8bf*)(p + 16);
  v16bf r;
#pragma unroll
  for (int i = 0; i < 8; ++i) { r[i] = c0[i]; r[8 + i] = c1[i]; }
  return r;
}

// ============ Kernel 0: one-shot weight convert f32 -> bf16 ============
// Wb = [Wq | Wk | Wv | Wo], each D*D, read once from HBM.
__global__ __launch_bounds__(256) void cvt_w_kernel(
    const float* __restrict__ Wq, const float* __restrict__ Wk,
    const float* __restrict__ Wv, const float* __restrict__ Wo,
    bf16* __restrict__ out) {
  const size_t i = ((size_t)blockIdx.x * 256 + threadIdx.x) * 8;  // over 4*D*D
  const int which = (int)(i >> 20);                               // D*D = 1Mi
  const float* src = (which == 0) ? Wq : (which == 1) ? Wk : (which == 2) ? Wv : Wo;
  const v8f x = *(const v8f*)(src + (i & ((1u << 20) - 1)));
  v8bf y;
#pragma unroll
  for (int j = 0; j < 8; ++j) y[j] = (bf16)x[j];
  *(v8bf*)(out + i) = y;
}

// ================= Kernel 1: fused QKV projection =================
// y = x @ W^T ; wave computes 32(M) x 64(N); 4 waves -> 128-row tile.
// Q,K -> [b][h][s][dk] bf16 ; V -> transposed [b][h][dk][s] bf16.
__global__ __launch_bounds__(128) void qkv_proj_kernel(
    const float* __restrict__ q, const float* __restrict__ k, const float* __restrict__ v,
    const bf16* __restrict__ Wb,
    bf16* __restrict__ Qh, bf16* __restrict__ Kh, bf16* __restrict__ Vt) {
  const int mode = blockIdx.z;
  const float* X = (mode == 0) ? q : (mode == 1) ? k : v;
  const bf16* W = Wb + (size_t)mode * D_ * D_;
  const int w = threadIdx.x >> 5, lane = threadIdx.x & 31;
  const int hl = lane >> 4, col = lane & 15;
  const int m0 = blockIdx.x * 128 + w * 32;  // wave's 32 rows of M
  const int n0 = blockIdx.y * 64;            // 64 cols, 4 tiles

  v8f acc[2][4];
#pragma unroll
  for (int mi = 0; mi < 2; ++mi)
#pragma unroll
    for (int t = 0; t < 4; ++t)
#pragma unroll
      for (int i = 0; i < 8; ++i) acc[mi][t][i] = 0.0f;

  for (int kk = 0; kk < D_ / 32; ++kk) {
    const v16bf a0 = load_a_f32(X + (size_t)m0 * D_, D_, kk * 32);
    const v16bf a1 = load_a_f32(X + (size_t)(m0 + 16) * D_, D_, kk * 32);
#pragma unroll
    for (int t = 0; t < 4; ++t) {
      const v16bf b = load_bt_bf16(W, D_, n0 + t * 16, kk * 32);
      acc[0][t] = wmma_bf16(a0, b, acc[0][t]);
      acc[1][t] = wmma_bf16(a1, b, acc[1][t]);
    }
  }

#pragma unroll
  for (int mi = 0; mi < 2; ++mi)
#pragma unroll
    for (int t = 0; t < 4; ++t)
#pragma unroll
      for (int r = 0; r < 8; ++r) {
        const int m = m0 + mi * 16 + r + 8 * hl;   // C layout: row = r+8*hl, col = lane&15
        const int bb = m >> 11, s = m & (S_ - 1);
        const int n = n0 + t * 16 + col;
        const int h = n >> 6, dk = n & (DK_ - 1);
        const bf16 val = (bf16)acc[mi][t][r];
        if (mode == 2) {
          Vt[((size_t)(bb * H_ + h) * DK_ + dk) * S_ + s] = val;
        } else {
          bf16* dst = (mode == 0) ? Qh : Kh;
          dst[((size_t)(bb * H_ + h) * S_ + s) * DK_ + dk] = val;
        }
      }
}

// ================= Kernel 2: causal flash attention =================
// Grid: (S/64, B*H). 4 waves/WG; wave owns 16 query rows x DK=64.
__global__ __launch_bounds__(128) void flash_attn_kernel(
    const bf16* __restrict__ Qh, const bf16* __restrict__ Kh,
    const bf16* __restrict__ Vt, bf16* __restrict__ Obuf) {
  __shared__ bf16 Plds[4][16][64];   // per-wave P staging (C-layout -> A-layout)
  const int w = threadIdx.x >> 5, lane = threadIdx.x & 31;
  const int hl = lane >> 4, col = lane & 15;
  const int bh = blockIdx.y;
  const int b = bh >> 4, h = bh & (H_ - 1);
  const int q0 = blockIdx.x * 64 + w * 16;

  const bf16* Qbase = Qh + ((size_t)bh * S_ + q0) * DK_;
  const bf16* Kbase = Kh + (size_t)bh * S_ * DK_;
  const bf16* Vbase = Vt + (size_t)bh * DK_ * S_;

  v16bf qf[2];
#pragma unroll
  for (int kk = 0; kk < 2; ++kk) qf[kk] = load_a_bf16(Qbase, DK_, kk * 32);

  v8f o[4];
  float m_i[8], l_i[8];
#pragma unroll
  for (int t = 0; t < 4; ++t)
#pragma unroll
    for (int i = 0; i < 8; ++i) o[t][i] = 0.0f;
#pragma unroll
  for (int r = 0; r < 8; ++r) { m_i[r] = -3.0e38f; l_i[r] = 0.0f; }

  const int nblocks = (q0 + 15) / 64 + 1;
  for (int ib = 0; ib < nblocks; ++ib) {
    const int kb = ib * 64;

    // prefetch next key block (K rows are 128B lines; V^T rows stride S_)
    if (ib + 1 < nblocks) {
      const int kn = kb + 64;
      __builtin_prefetch(Kbase + (size_t)(kn + lane * 2) * DK_, 0, 1);
      __builtin_prefetch(Vbase + (size_t)(lane * 2) * S_ + kn, 0, 1);
      __builtin_prefetch(Vbase + (size_t)(lane * 2 + 1) * S_ + kn, 0, 1);
    }

    // scores = Q @ K^T  (8 WMMAs)
    v8f sc[4];
#pragma unroll
    for (int t = 0; t < 4; ++t)
#pragma unroll
      for (int i = 0; i < 8; ++i) sc[t][i] = 0.0f;
#pragma unroll
    for (int kk = 0; kk < 2; ++kk)
#pragma unroll
      for (int t = 0; t < 4; ++t)
        sc[t] = wmma_bf16(qf[kk], load_bt_bf16(Kbase, DK_, kb + t * 16, kk * 32), sc[t]);

    // scale into base-2 domain + causal mask (analytic)
#pragma unroll
    for (int t = 0; t < 4; ++t)
#pragma unroll
      for (int r = 0; r < 8; ++r) {
        const float sv = sc[t][r] * SSCALE;
        const int qg = q0 + r + 8 * hl;
        const int kg = kb + t * 16 + col;
        sc[t][r] = (kg <= qg) ? sv : NEGV;
      }

    // online softmax (base 2): row lives across 16 lanes of one half
    float alpha[8];
#pragma unroll
    for (int r = 0; r < 8; ++r) {
      float mx = fmaxf(fmaxf(sc[0][r], sc[1][r]), fmaxf(sc[2][r], sc[3][r]));
#pragma unroll
      for (int off = 8; off >= 1; off >>= 1) mx = fmaxf(mx, __shfl_xor(mx, off, 32));
      const float mnew = fmaxf(m_i[r], mx);
      alpha[r] = exp2f(m_i[r] - mnew);
      m_i[r] = mnew;
      float rs = 0.0f;
#pragma unroll
      for (int t = 0; t < 4; ++t) {
        const float p = exp2f(sc[t][r] - mnew);
        sc[t][r] = p;
        rs += p;
      }
#pragma unroll
      for (int off = 8; off >= 1; off >>= 1) rs += __shfl_xor(rs, off, 32);
      l_i[r] = l_i[r] * alpha[r] + rs;
#pragma unroll
      for (int t = 0; t < 4; ++t) o[t][r] *= alpha[r];
    }

    // stage P into per-wave LDS [16][64]; same-wave DS ops are in order
#pragma unroll
    for (int t = 0; t < 4; ++t)
#pragma unroll
      for (int r = 0; r < 8; ++r)
        Plds[w][r + 8 * hl][t * 16 + col] = (bf16)sc[t][r];

    // O += P @ V  (8 WMMAs; V^T rows contiguous in key dim)
#pragma unroll
    for (int kk = 0; kk < 2; ++kk) {
      const v16bf pa = load_a_bf16(&Plds[w][0][0], 64, kk * 32);
#pragma unroll
      for (int t = 0; t < 4; ++t)
        o[t] = wmma_bf16(pa, load_bt_bf16(Vbase, S_, t * 16, kb + kk * 32), o[t]);
    }
  }

  // normalize and store merged-head layout [b][s][h*DK+dk]
#pragma unroll
  for (int r = 0; r < 8; ++r) {
    const float inv = 1.0f / l_i[r];
    const int s = q0 + r + 8 * hl;
#pragma unroll
    for (int t = 0; t < 4; ++t)
      Obuf[(size_t)(b * S_ + s) * D_ + h * DK_ + t * 16 + col] = (bf16)(o[t][r] * inv);
  }
}

// ================= Kernel 3: output projection =================
__global__ __launch_bounds__(128) void out_proj_kernel(
    const bf16* __restrict__ Obuf, const bf16* __restrict__ WoB, float* __restrict__ Out) {
  const int w = threadIdx.x >> 5, lane = threadIdx.x & 31;
  const int hl = lane >> 4, col = lane & 15;
  const int m0 = blockIdx.x * 128 + w * 32;
  const int n0 = blockIdx.y * 64;

  v8f acc[2][4];
#pragma unroll
  for (int mi = 0; mi < 2; ++mi)
#pragma unroll
    for (int t = 0; t < 4; ++t)
#pragma unroll
      for (int i = 0; i < 8; ++i) acc[mi][t][i] = 0.0f;

  for (int kk = 0; kk < D_ / 32; ++kk) {
    const v16bf a0 = load_a_bf16(Obuf + (size_t)m0 * D_, D_, kk * 32);
    const v16bf a1 = load_a_bf16(Obuf + (size_t)(m0 + 16) * D_, D_, kk * 32);
#pragma unroll
    for (int t = 0; t < 4; ++t) {
      const v16bf b = load_bt_bf16(WoB, D_, n0 + t * 16, kk * 32);
      acc[0][t] = wmma_bf16(a0, b, acc[0][t]);
      acc[1][t] = wmma_bf16(a1, b, acc[1][t]);
    }
  }

#pragma unroll
  for (int mi = 0; mi < 2; ++mi)
#pragma unroll
    for (int t = 0; t < 4; ++t)
#pragma unroll
      for (int r = 0; r < 8; ++r)
        Out[(size_t)(m0 + mi * 16 + r + 8 * hl) * D_ + n0 + t * 16 + col] = acc[mi][t][r];
}

extern "C" void kernel_launch(void* const* d_in, const int* in_sizes, int n_in,
                              void* d_out, int out_size, void* d_ws, size_t ws_size,
                              hipStream_t stream) {
  const float* q  = (const float*)d_in[0];
  const float* k  = (const float*)d_in[1];
  const float* v  = (const float*)d_in[2];
  // d_in[3] = causal mask: applied analytically in the flash kernel
  const float* Wq = (const float*)d_in[4];
  const float* Wk = (const float*)d_in[5];
  const float* Wv = (const float*)d_in[6];
  const float* Wo = (const float*)d_in[7];

  const size_t headElems = (size_t)B_ * H_ * S_ * DK_;   // 4Mi elems
  bf16* Qh   = (bf16*)d_ws;
  bf16* Kh   = Qh + headElems;
  bf16* Vt   = Kh + headElems;
  bf16* Obuf = Vt + headElems;
  bf16* Wb   = Obuf + headElems;                          // 4 x D*D bf16 (8 MB)

  const dim3 blk128(128, 1, 1);
  cvt_w_kernel<<<dim3((4 * D_ * D_) / (8 * 256)), dim3(256), 0, stream>>>(Wq, Wk, Wv, Wo, Wb);
  qkv_proj_kernel<<<dim3((B_ * S_) / 128, D_ / 64, 3), blk128, 0, stream>>>(
      q, k, v, Wb, Qh, Kh, Vt);
  flash_attn_kernel<<<dim3(S_ / 64, B_ * H_), blk128, 0, stream>>>(Qh, Kh, Vt, Obuf);
  out_proj_kernel<<<dim3((B_ * S_) / 128, D_ / 64), blk128, 0, stream>>>(
      Obuf, Wb + (size_t)3 * D_ * D_, (float*)d_out);
}
